// LlamaAttention_3685081940528
// MI455X (gfx1250) — matrix operand order repaired and checked by reading, not compiled, not run on hardware
//
#include <hip/hip_runtime.h>
#include <hip/hip_bf16.h>

// ---------------------------------------------------------------------------
// Types for CDNA5 WMMA (wave32): 16x16x32 bf16 -> f32 accumulate
// ---------------------------------------------------------------------------
typedef __attribute__((ext_vector_type(16))) __bf16 bf16x16;
typedef __attribute__((ext_vector_type(8)))  float  f32x8;
typedef __attribute__((ext_vector_type(4)))  unsigned int u32x4;
typedef __attribute__((ext_vector_type(8)))  int    i32x8;
typedef __attribute__((ext_vector_type(4)))  int    i32x4;

union Frag {
    bf16x16 v;
    uint4   u[2];
};

__device__ __forceinline__ f32x8 wmma_bf16(const Frag& a, const Frag& b, f32x8 c) {
    return __builtin_amdgcn_wmma_f32_16x16x32_bf16(false, a.v, false, b.v,
                                                   (short)0, c, false, false);
}

static constexpr int S_LEN  = 2048;
static constexpr int HID    = 4096;
static constexpr int NH     = 32;
static constexpr int NKV    = 8;
static constexpr int HD     = 128;
static constexpr float SM_SCALE = 0.08838834764831845f; // 1/sqrt(128)
static constexpr float NEG_BIG  = -3.0e38f;

// TDM availability (device pass only; host pass uses fallback copy path)
#if defined(__HIP_DEVICE_COMPILE__) && __has_builtin(__builtin_amdgcn_tensor_load_to_lds)
#define USE_TDM 1
#else
#define USE_TDM 0
#endif

// ---------------------------------------------------------------------------
// fp32 -> bf16 elementwise cast
// ---------------------------------------------------------------------------
__global__ void cast_bf16(const float* __restrict__ x,
                          __hip_bfloat16* __restrict__ y, size_t n) {
    size_t i = (size_t)blockIdx.x * blockDim.x + threadIdx.x;
    size_t stride = (size_t)gridDim.x * blockDim.x;
    for (; i < n; i += stride) y[i] = __float2bfloat16(x[i]);
}

// ---------------------------------------------------------------------------
// C[M,N] (f32) = A[M,K] (bf16, row-major) * B[N,K]^T (bf16, row-major)
// Block: 256 threads = 8 waves in a 2(M) x 4(N) grid; each wave: 32x64 tile
// via a 2x4 grid of 16x16 WMMA accumulators. K streamed in chunks of 32.
// ---------------------------------------------------------------------------
__global__ __launch_bounds__(256)
void gemm_bf16_nt(const __hip_bfloat16* __restrict__ A,
                  const __hip_bfloat16* __restrict__ B,
                  float* __restrict__ C, int M, int N, int K) {
    const int lane = threadIdx.x & 31;
    const int wave = threadIdx.x >> 5;
    const int wm   = wave & 1;         // 0..1
    const int wn   = wave >> 1;        // 0..3
    const int half = lane >> 4;        // lane>=16
    const int l16  = lane & 15;

    const int mBase = blockIdx.y * 64  + wm * 32;
    const int nBase = blockIdx.x * 256 + wn * 64;

    f32x8 acc[2][4];
#pragma unroll
    for (int i = 0; i < 2; ++i)
#pragma unroll
        for (int j = 0; j < 4; ++j) acc[i][j] = f32x8{};

    const __hip_bfloat16* aRow0 = A + (size_t)(mBase +      l16) * K;
    const __hip_bfloat16* aRow1 = A + (size_t)(mBase + 16 + l16) * K;
    const __hip_bfloat16* bRow[4];
#pragma unroll
    for (int j = 0; j < 4; ++j)
        bRow[j] = B + (size_t)(nBase + j * 16 + l16) * K;

    const int aOff = half * 8;   // A: chunks at aOff and aOff+16
    const int bOff = half * 8;   // B: chunks at bOff and bOff+16

    for (int k = 0; k < K; k += 32) {
        // speculative prefetch of next tiles (global_prefetch_b8)
        __builtin_prefetch(aRow0 + k + 256, 0, 1);
        __builtin_prefetch(bRow[0] + k + 256, 0, 1);

        Frag a[2], b[4];
        a[0].u[0] = *(const uint4*)(aRow0 + k + aOff);
        a[0].u[1] = *(const uint4*)(aRow0 + k + aOff + 16);
        a[1].u[0] = *(const uint4*)(aRow1 + k + aOff);
        a[1].u[1] = *(const uint4*)(aRow1 + k + aOff + 16);
#pragma unroll
        for (int j = 0; j < 4; ++j) {
            b[j].u[0] = *(const uint4*)(bRow[j] + k + bOff);
            b[j].u[1] = *(const uint4*)(bRow[j] + k + bOff + 16);
        }
#pragma unroll
        for (int i = 0; i < 2; ++i)
#pragma unroll
            for (int j = 0; j < 4; ++j)
                acc[i][j] = wmma_bf16(a[i], b[j], acc[i][j]);
    }

    // C layout: VGPR e, lane -> row = e + 8*half, col = l16
#pragma unroll
    for (int i = 0; i < 2; ++i) {
        const int r0 = mBase + i * 16 + half * 8;
#pragma unroll
        for (int j = 0; j < 4; ++j) {
            const int c0 = nBase + j * 16 + l16;
#pragma unroll
            for (int e = 0; e < 8; ++e)
                C[(size_t)(r0 + e) * N + c0] = acc[i][j][e];
        }
    }
}

// ---------------------------------------------------------------------------
// RoPE on f32 projections, emit bf16. One thread per (s, head, freq-pair j).
// ---------------------------------------------------------------------------
__global__ void rope_cast(const float* __restrict__ X, const int* __restrict__ pos,
                          __hip_bfloat16* __restrict__ Y, int nHeads) {
    size_t idx   = (size_t)blockIdx.x * blockDim.x + threadIdx.x;
    size_t total = (size_t)S_LEN * nHeads * 64;
    if (idx >= total) return;
    int j  = (int)(idx & 63);
    size_t t = idx >> 6;
    int hh = (int)(t % nHeads);
    int s  = (int)(t / nHeads);
    float p   = (float)pos[s];
    float inv = __powf(10000.0f, -(float)(2 * j) / 128.0f);
    float ang = p * inv;
    float c = __cosf(ang), sn = __sinf(ang);
    size_t base = (size_t)s * (nHeads * HD) + (size_t)hh * HD + j;
    float x1 = X[base], x2 = X[base + 64];
    Y[base]      = __float2bfloat16(x1 * c - x2 * sn);
    Y[base + 64] = __float2bfloat16(x2 * c + x1 * sn);
}

// ---------------------------------------------------------------------------
// V (f32 [s, NKV*HD]) -> Vt (bf16 [NKV*HD, S]) so PV B-fragments are contiguous
// ---------------------------------------------------------------------------
__global__ void v_transpose_cast(const float* __restrict__ Vf,
                                 __hip_bfloat16* __restrict__ Vt) {
    size_t idx = (size_t)blockIdx.x * blockDim.x + threadIdx.x;
    size_t total = (size_t)S_LEN * NKV * HD;
    if (idx >= total) return;
    int d = (int)(idx % (NKV * HD));
    int s = (int)(idx / (NKV * HD));
    Vt[(size_t)d * S_LEN + s] = __float2bfloat16(Vf[idx]);
}

// ---------------------------------------------------------------------------
// Tensor DMA descriptor builder (ISA cdna5 8.3/8.4): 2-D tile, bf16 elements.
// All inputs are wave-uniform -> descriptor materializes into SGPR groups.
//   lds    : LDS byte offset of tile destination
//   gaddr  : global byte address of tile start
//   dim0/1 : tensor dims (elements); tdim0/1: tile dims (elements)
//   stride0: tensor dim0 stride (elements)
// ---------------------------------------------------------------------------
#if USE_TDM
__device__ __forceinline__ void tdm_desc_2d(unsigned lds, unsigned long long gaddr,
                                            unsigned dim0, unsigned dim1,
                                            unsigned tdim0, unsigned tdim1,
                                            unsigned long long stride0,
                                            u32x4& g0, i32x8& g1) {
    g0[0] = 1u;                                             // count=1, user D#
    g0[1] = lds;                                            // lds_addr [63:32]
    g0[2] = (unsigned)(gaddr & 0xFFFFFFFFull);              // global_addr lo
    g0[3] = (unsigned)((gaddr >> 32) & 0x01FFFFFFull)
          | 0x80000000u;                                    // addr hi | type=2
    g1[0] = (int)(1u << 16);                                // data_size=2B, mask=0
    g1[1] = (int)((dim0 & 0xFFFFu) << 16);                  // tensor_dim0 lo
    g1[2] = (int)((dim0 >> 16) | ((dim1 & 0xFFFFu) << 16)); // dim0 hi | dim1 lo
    g1[3] = (int)((dim1 >> 16) | (tdim0 << 16));            // dim1 hi | tile_dim0
    g1[4] = (int)(tdim1 & 0xFFFFu);                         // tile_dim1 (tile_dim2=0)
    g1[5] = (int)(stride0 & 0xFFFFFFFFull);                 // stride0 lo
    g1[6] = (int)((stride0 >> 32) & 0xFFFFull);             // stride0 hi (stride1=0)
    g1[7] = 0;
}

__device__ __forceinline__ void tdm_load(const u32x4& g0, const i32x8& g1) {
    i32x4 z4 = {0, 0, 0, 0};
#if defined(__clang_major__) && __clang_major__ >= 23
    i32x8 z8 = {0, 0, 0, 0, 0, 0, 0, 0};
    __builtin_amdgcn_tensor_load_to_lds(g0, g1, z4, z4, z8, 0);
#else
    __builtin_amdgcn_tensor_load_to_lds(g0, g1, z4, z4, 0);
#endif
}
#endif // USE_TDM

// ---------------------------------------------------------------------------
// Causal flash attention with online softmax, block-cooperative K/V staging.
// Grid: (S/64 q-tiles, NH heads). Block: 128 threads = 4 waves; wave w owns
// q rows [qTile + 16w, +16). Key blocks of 32 (== WMMA K; 2 N-tiles of 16 in QK^T).
// Wave 0 issues TDM loads of the K tile (32x128) and V tile (128x32, from the
// pre-transposed Vt) into LDS; s_wait_tensorcnt + barrier publish them.
// QK^T and PV both on v_wmma_f32_16x16x32_bf16; P goes through LDS to be
// re-laid-out from C-accumulator layout into A-fragment layout.
// ---------------------------------------------------------------------------
__global__ __launch_bounds__(128)
void flash_attn(const __hip_bfloat16* __restrict__ Qr,  // [S, NH*HD]
                const __hip_bfloat16* __restrict__ Kr,  // [S, NKV*HD]
                const __hip_bfloat16* __restrict__ Vt,  // [NKV*HD, S]
                __hip_bfloat16* __restrict__ O) {       // [S, NH*HD]
    const int lane  = threadIdx.x & 31;
    const int wave  = threadIdx.x >> 5;      // 0..3
    const int half  = lane >> 4;
    const int l16   = lane & 15;
    const int h     = blockIdx.y;
    const int hk    = h >> 2;                // GQA: 4 Q heads per KV head
    const int qTile = blockIdx.x * 64;
    const int qw    = qTile + wave * 16;     // this wave's first q row

    __shared__ __align__(16) __hip_bfloat16 Kt[32][HD];   // [key][d]   8 KB
    __shared__ __align__(16) __hip_bfloat16 Vs[HD][32];   // [d][key]   8 KB
    __shared__ __align__(16) __hip_bfloat16 Plds[4][16 * 32];

    // Preload Q A-fragments: K-dim (head_dim=128) = 4 chunks of 32
    Frag qf[4];
    const __hip_bfloat16* qRow = Qr + (size_t)(qw + l16) * (NH * HD) + (size_t)h * HD;
#pragma unroll
    for (int kk = 0; kk < 4; ++kk) {
        qf[kk].u[0] = *(const uint4*)(qRow + kk * 32 + half * 8);
        qf[kk].u[1] = *(const uint4*)(qRow + kk * 32 + half * 8 + 16);
    }

    float m[8], l[8];
    f32x8 accO[8];
#pragma unroll
    for (int e = 0; e < 8; ++e) { m[e] = NEG_BIG; l[e] = 0.f; }
#pragma unroll
    for (int n = 0; n < 8; ++n) accO[n] = f32x8{};

    const int kbMaxBlk = (qTile + 63) >> 5;  // inclusive, causal (block range)
    for (int kb = 0; kb <= kbMaxBlk; ++kb) {
        const int k0 = kb * 32;

        __syncthreads();   // previous tile fully consumed before overwrite
#if USE_TDM
        if (wave == 0) {
            // K tile: tensor Kr [S, NKV*HD]; tile 32 keys x 128 d -> Kt[32][128]
            u32x4 g0; i32x8 g1;
            tdm_desc_2d((unsigned)(size_t)&Kt[0][0],
                        (unsigned long long)(size_t)(Kr + (size_t)k0 * (NKV * HD) + (size_t)hk * HD),
                        NKV * HD, S_LEN, HD, 32, NKV * HD, g0, g1);
            tdm_load(g0, g1);
            // V tile: tensor Vt [NKV*HD, S]; tile 128 d x 32 keys -> Vs[128][32]
            tdm_desc_2d((unsigned)(size_t)&Vs[0][0],
                        (unsigned long long)(size_t)(Vt + (size_t)hk * HD * S_LEN + k0),
                        S_LEN, NKV * HD, 32, HD, S_LEN, g0, g1);
            tdm_load(g0, g1);
            __builtin_amdgcn_s_wait_tensorcnt(0);
        }
#else
        {   // cooperative fallback copy: 512 uint4 per tile / 128 threads
            const int t = threadIdx.x;
#pragma unroll
            for (int i = 0; i < 4; ++i) {
                const int idx = t * 4 + i;                 // 0..511
                const int kr = idx >> 4, kc = (idx & 15) * 8;
                *(uint4*)&Kt[kr][kc] =
                    *(const uint4*)(Kr + (size_t)(k0 + kr) * (NKV * HD) + hk * HD + kc);
                const int vr = idx >> 2, vc = (idx & 3) * 8;
                *(uint4*)&Vs[vr][vc] =
                    *(const uint4*)(Vt + (size_t)(hk * HD + vr) * S_LEN + k0 + vc);
            }
        }
#endif
        __syncthreads();   // tiles visible to all waves

        if (k0 > qw + 15) continue;   // wave-uniform causal skip (EXEC stays full)

        // ---- scores: 16 q-rows x 32 keys = 2 accumulators (K frags from LDS)
        f32x8 sc[2] = {f32x8{}, f32x8{}};
#pragma unroll
        for (int t = 0; t < 2; ++t) {
            const __hip_bfloat16* kRow = &Kt[t * 16 + l16][0];
#pragma unroll
            for (int kk = 0; kk < 4; ++kk) {
                Frag bf;
                bf.u[0] = *(const uint4*)(kRow + kk * 32 + half * 8);
                bf.u[1] = *(const uint4*)(kRow + kk * 32 + half * 8 + 16);
                sc[t] = wmma_bf16(qf[kk], bf, sc[t]);
            }
        }

        // ---- scale + causal mask; per-lane row = qw + e + 8*half
        float sv[2][8];
#pragma unroll
        for (int t = 0; t < 2; ++t) {
            const int kcol = k0 + t * 16 + l16;
#pragma unroll
            for (int e = 0; e < 8; ++e) {
                const int qrow = qw + e + 8 * half;
                sv[t][e] = (kcol > qrow) ? NEG_BIG : sc[t][e] * SM_SCALE;
            }
        }

        // ---- row max over 32 keys (2 regs + 16-lane shuffle reduction)
        float rmax[8];
#pragma unroll
        for (int e = 0; e < 8; ++e) rmax[e] = fmaxf(sv[0][e], sv[1][e]);
#pragma unroll
        for (int off = 1; off < 16; off <<= 1)
#pragma unroll
            for (int e = 0; e < 8; ++e)
                rmax[e] = fmaxf(rmax[e], __shfl_xor(rmax[e], off, 16));

        float mnew[8], resc[8], p0[8], p1[8], psum[8];
#pragma unroll
        for (int e = 0; e < 8; ++e) {
            mnew[e] = fmaxf(m[e], rmax[e]);
            resc[e] = __expf(m[e] - mnew[e]);
            p0[e]   = __expf(sv[0][e] - mnew[e]);
            p1[e]   = __expf(sv[1][e] - mnew[e]);
            psum[e] = p0[e] + p1[e];
        }
#pragma unroll
        for (int off = 1; off < 16; off <<= 1)
#pragma unroll
            for (int e = 0; e < 8; ++e)
                psum[e] += __shfl_xor(psum[e], off, 16);
#pragma unroll
        for (int e = 0; e < 8; ++e) {
            l[e] = l[e] * resc[e] + psum[e];
            m[e] = mnew[e];
        }
#pragma unroll
        for (int n = 0; n < 8; ++n)
#pragma unroll
            for (int e = 0; e < 8; ++e) accO[n][e] *= resc[e];

        // ---- P (C-layout) -> LDS row-major [16][32] (per-wave region)
        __hip_bfloat16* pw = &Plds[wave][0];
#pragma unroll
        for (int e = 0; e < 8; ++e) {
            const int r = e + 8 * half;
            pw[r * 32 +      l16] = __float2bfloat16(p0[e]);
            pw[r * 32 + 16 + l16] = __float2bfloat16(p1[e]);
        }
        asm volatile("s_wait_dscnt 0" ::: "memory");

        // ---- reload P as A-fragment 16x32
        Frag pa;
        const __hip_bfloat16* pr = pw + l16 * 32;
        pa.u[0] = *(const uint4*)(pr + half * 8);
        pa.u[1] = *(const uint4*)(pr + half * 8 + 16);

        // ---- PV: B column = head-dim, K = 32 keys (V frags from LDS)
#pragma unroll
        for (int nt = 0; nt < 8; ++nt) {
            const __hip_bfloat16* vRow = &Vs[nt * 16 + l16][0] + half * 8;
            Frag vb;
            vb.u[0] = *(const uint4*)(vRow);
            vb.u[1] = *(const uint4*)(vRow + 16);
            accO[nt] = wmma_bf16(pa, vb, accO[nt]);
        }
    }

    // ---- epilogue: normalize and store bf16 [s, NH*HD]
#pragma unroll
    for (int nt = 0; nt < 8; ++nt) {
        const int c = h * HD + nt * 16 + l16;
#pragma unroll
        for (int e = 0; e < 8; ++e) {
            const int s = qw + e + 8 * half;
            O[(size_t)s * (NH * HD) + c] = __float2bfloat16(accO[nt][e] / l[e]);
        }
    }
}

// ---------------------------------------------------------------------------
// Launch
// ---------------------------------------------------------------------------
extern "C" void kernel_launch(void* const* d_in, const int* in_sizes, int n_in,
                              void* d_out, int out_size, void* d_ws, size_t ws_size,
                              hipStream_t stream) {
    (void)in_sizes; (void)n_in; (void)out_size; (void)ws_size;
    const float* h   = (const float*)d_in[0];
    const int*   pos = (const int*)d_in[1];
    const float* Wq  = (const float*)d_in[2];
    const float* Wk  = (const float*)d_in[3];
    const float* Wv  = (const float*)d_in[4];
    const float* Wo  = (const float*)d_in[5];
    float* out = (float*)d_out;

    char*  ws  = (char*)d_ws;
    size_t off = 0;
    auto alloc = [&](size_t bytes) { char* p = ws + off; off += (bytes + 255) & ~(size_t)255; return p; };

    const size_t nH  = (size_t)S_LEN * HID;      // 8.4M
    const size_t nWq = (size_t)HID * HID;        // 16.8M
    const size_t nWk = (size_t)(NKV * HD) * HID; // 4.2M
    const size_t nKV = (size_t)S_LEN * NKV * HD; // 2.1M

    __hip_bfloat16* hB  = (__hip_bfloat16*)alloc(nH  * 2);
    __hip_bfloat16* WqB = (__hip_bfloat16*)alloc(nWq * 2);
    __hip_bfloat16* WkB = (__hip_bfloat16*)alloc(nWk * 2);
    __hip_bfloat16* WvB = (__hip_bfloat16*)alloc(nWk * 2);
    __hip_bfloat16* WoB = (__hip_bfloat16*)alloc(nWq * 2);
    float*          Qf  = (float*)alloc(nH  * 4);
    float*          Kf  = (float*)alloc(nKV * 4);
    float*          Vf  = (float*)alloc(nKV * 4);
    __hip_bfloat16* QrB = (__hip_bfloat16*)alloc(nH  * 2);
    __hip_bfloat16* KrB = (__hip_bfloat16*)alloc(nKV * 2);
    __hip_bfloat16* VtB = (__hip_bfloat16*)alloc(nKV * 2);
    __hip_bfloat16* AoB = (__hip_bfloat16*)alloc(nH  * 2);

    // 1) fp32 -> bf16 casts
    cast_bf16<<<2048, 256, 0, stream>>>(h,  hB,  nH);
    cast_bf16<<<2048, 256, 0, stream>>>(Wq, WqB, nWq);
    cast_bf16<<<2048, 256, 0, stream>>>(Wk, WkB, nWk);
    cast_bf16<<<2048, 256, 0, stream>>>(Wv, WvB, nWk);
    cast_bf16<<<2048, 256, 0, stream>>>(Wo, WoB, nWq);

    // 2) QKV projections (NT GEMM, f32 accum)
    gemm_bf16_nt<<<dim3(HID / 256, S_LEN / 64), 256, 0, stream>>>(hB, WqB, Qf, S_LEN, HID, HID);
    gemm_bf16_nt<<<dim3((NKV * HD) / 256, S_LEN / 64), 256, 0, stream>>>(hB, WkB, Kf, S_LEN, NKV * HD, HID);
    gemm_bf16_nt<<<dim3((NKV * HD) / 256, S_LEN / 64), 256, 0, stream>>>(hB, WvB, Vf, S_LEN, NKV * HD, HID);

    // 3) RoPE (+ cast to bf16)
    {
        size_t tq = (size_t)S_LEN * NH  * 64;
        size_t tk = (size_t)S_LEN * NKV * 64;
        rope_cast<<<(unsigned)((tq + 255) / 256), 256, 0, stream>>>(Qf, pos, QrB, NH);
        rope_cast<<<(unsigned)((tk + 255) / 256), 256, 0, stream>>>(Kf, pos, KrB, NKV);
    }

    // 4) V transpose + cast
    v_transpose_cast<<<(unsigned)((nKV + 255) / 256), 256, 0, stream>>>(Vf, VtB);

    // 5) causal flash attention (TDM-staged K/V tiles)
    flash_attn<<<dim3(S_LEN / 64, NH), 128, 0, stream>>>(QrB, KrB, VtB, AoB);

    // 6) output projection -> f32 d_out
    gemm_bf16_nt<<<dim3(HID / 256, S_LEN / 64), 256, 0, stream>>>(AoB, WoB, out, S_LEN, HID, HID);
}
